// CarafeLayer_23115513987460
// MI455X (gfx1250) — matrix operand order repaired and checked by reading, not compiled, not run on hardware
//
#include <hip/hip_runtime.h>

typedef __attribute__((ext_vector_type(16))) _Float16 v16h;
typedef __attribute__((ext_vector_type(4)))  _Float16 v4h;
typedef __attribute__((ext_vector_type(8)))  float    v8f;
typedef __attribute__((ext_vector_type(4)))  float    v4f;
typedef __attribute__((ext_vector_type(8)))  unsigned int v8u;

#define KS   5          // reassembly kernel size
#define KK2  25         // KS*KS taps
#define UPF  2          // upsample factor (UP*UP = 4 sub-pixels)
#define CB   256        // channels
#define HB   64
#define WB   64
#define PIX  8          // pixels (along W) per workgroup
#define TWW  16         // staged window width: cols wcol0-4 .. wcol0+11
#define XST  84         // per-channel stride in halfs: 4B-aligned spans, no bank conflicts
#define OGS  68         // ostage group stride (floats): distinct banks per group
#define THREADS 256

// Element e (0..15) of half-wave kh -> tap (i*5+j), or -1 (zero pad in B).
// Rows are pair-aligned so every A register pair lives in one row span.
__host__ __device__ constexpr int tap_of_elem(int kh, int e) {
    if (kh == 0) {
        if (e <= 4)  return 0 * KS + e;          // row0 j0..4
        if (e == 5)  return -1;
        if (e <= 10) return 1 * KS + (e - 6);    // row1 j0..4
        if (e == 11) return -1;
        return 2 * KS + (e - 12);                // row2 j0..3
    } else {
        if (e <= 4)  return 3 * KS + e;          // row3 j0..4
        if (e == 5)  return -1;
        if (e <= 10) return 4 * KS + (e - 6);    // row4 j0..4
        if (e == 11) return -1;
        if (e == 12) return 2 * KS + 4;          // row2 j4
        return -1;
    }
}
// Invert the 16-bit A-layout K-slot mapping: K = e + (e&8) + 8*kh.
__host__ __device__ constexpr int tap_of_K(int K) {
    int kh, e;
    if      (K <  8) { kh = 0; e = K; }
    else if (K < 16) { kh = 1; e = K - 8; }
    else if (K < 24) { kh = 0; e = K - 8; }
    else             { kh = 1; e = K - 16; }
    return tap_of_elem(kh, e);
}

// pair pick from a 7-dword row span: even offset = mov, odd = one v_alignbit_b32
#define PICK(S, o) (((o) & 1) ? __builtin_amdgcn_alignbit(S[((o) >> 1) + 1], S[(o) >> 1], 16) \
                              : S[(o) >> 1])

__global__ __launch_bounds__(THREADS)
void carafe_wmma_kernel(const float* __restrict__ x,
                        const float* __restrict__ km,
                        float* __restrict__ out)
{
    __shared__ __align__(16) _Float16 xtile[CB * XST];   // 43008 B halo tile (f16)
    union SmemU {                                        // build-phase buffers alias ostage
        struct {
            __align__(32) _Float16 bfrag[PIX * 32 * 16]; // B frags, lane-major (8 KB)
            _Float16 kmt[PIX * 104];                     // raw kernel-map taps (1.6 KB)
        } build;
        float ostage[8][10 * OGS];                       // 8 live + 2 dump groups/wave
    };
    __shared__ SmemU u;

    const int tid   = threadIdx.x;
    const int wave  = tid >> 5;
    const int lane  = tid & 31;
    const int col   = lane & 15;     // B/C/D column (N); u lives in cols 0..3
    const int khalf = lane >> 4;

    int g = blockIdx.x;
    const int wcol0 = (g & (WB / PIX - 1)) * PIX;
    g /= (WB / PIX);
    const int h = g & (HB - 1);
    const int b = g / HB;

    // ---- stage 5x16 halo window per channel: uniform float4 loads, quad-granular
    //      zero padding (h rows and the first/last quad at the W edges). ----
    const long xbase = (long)b * CB * (HB * WB);
    for (int it = 0; it < 20; ++it) {        // 5120 quad-tasks / 256 threads
        int s   = tid + it * THREADS;
        int q   = s & 3;
        int cr  = s >> 2;
        int c   = cr & (CB - 1);
        int r   = cr >> 8;                   // 0..4 halo row
        int gh  = h + r - 2;
        int gws = wcol0 - 4 + q * 4;
        bool inb = ((unsigned)gh < HB) & ((unsigned)gws <= WB - 4);
        long off = inb ? xbase + ((long)c * HB + gh) * WB + gws : xbase;
        v4f v = *(const v4f*)(x + off);      // 16B-aligned (wcol0 % 8 == 0)
        v4h hv;
        hv[0] = (_Float16)(inb ? v[0] : 0.f);
        hv[1] = (_Float16)(inb ? v[1] : 0.f);
        hv[2] = (_Float16)(inb ? v[2] : 0.f);
        hv[3] = (_Float16)(inb ? v[3] : 0.f);
        *(v4h*)&xtile[c * XST + r * TWW + q * 4] = hv;   // byte addr % 8 == 0
    }
    // ---- stage kernel map (100 taps x 8 pixels) as f16 ----
    const long kbase = (long)b * (UPF * UPF * KK2) * (HB * WB);
    for (int idx = tid; idx < PIX * (UPF * UPF * KK2); idx += THREADS) {
        int p = idx & (PIX - 1);
        int q = idx >> 3;
        u.build.kmt[p * 104 + q] =
            (_Float16)km[kbase + ((long)q * HB + h) * WB + wcol0 + p];
    }
    __syncthreads();

    // ---- build B fragments in exact wave layout; zeros at pads and cols>=4 ----
    {
        int p  = tid >> 5;                   // PIX*32 == THREADS
        int l  = tid & 31;
        int n  = l & 15;
        int kh = l >> 4;
        #pragma unroll
        for (int e = 0; e < 16; ++e) {
            const int t0 = tap_of_elem(0, e); // element e for kh==0
            const int t1 = tap_of_elem(1, e); // element e for kh==1
            int tap = kh ? t1 : t0;          // select of two constants
            _Float16 v = (_Float16)0.f;
            if (n < UPF * UPF && tap >= 0)
                v = u.build.kmt[p * 104 + n * KK2 + tap];
            u.build.bfrag[tid * 16 + e] = v;
        }
    }
    __syncthreads();

    // ---- preload all 8 B fragments to registers (one aligned 32B read each) ----
    v16h bfr[PIX];
    #pragma unroll
    for (int p = 0; p < PIX; ++p)
        bfr[p] = *(const v16h*)&u.build.bfrag[(p * 32 + lane) * 16];
    __syncthreads();   // build buffers dead; ostage may now alias them

    // per-lane span bases (half-index within channel); khalf=1 row-2 span is
    // shifted +4 so pick indices are identical compile-time constants for all lanes
    const int sb0 = (khalf ? 3 : 0) * TWW + 2;
    const int sb1 = (khalf ? 4 : 1) * TWW + 2;
    const int sb2 = 2 * TWW + 2 + (khalf ? 4 : 0);

    // branchless C scatter: group k = col*2+khalf (dead cols -> dump groups 8,9)
    const int ogrp = (col < UPF * UPF) ? (col * 2 + khalf) : (8 + khalf);
    float* op = &u.ostage[wave][ogrp * OGS];

    #pragma unroll
    for (int t = 0; t < 2; ++t) {                   // 2 channel tiles/wave -> 256 ch
        const int c0 = (wave * 2 + t) * 16;
        const _Float16* ch = &xtile[(c0 + col) * XST];

        // load the three 14-half row spans once: 21 aligned ds_load_b32, one wait
        unsigned int S0[7], S1[7], S2[7];
        const unsigned int* p0 = (const unsigned int*)(ch + sb0);
        const unsigned int* p1 = (const unsigned int*)(ch + sb1);
        const unsigned int* p2 = (const unsigned int*)(ch + sb2);
        #pragma unroll
        for (int i = 0; i < 7; ++i) { S0[i] = p0[i]; S1[i] = p1[i]; S2[i] = p2[i]; }

        #pragma unroll
        for (int p = 0; p < PIX; ++p) {
            // A fragment: 8 mov/alignbit picks from the sliding spans
            v8u au;
            au[0] = PICK(S0, p + 0);
            au[1] = PICK(S0, p + 2);
            au[2] = PICK(S0, p + 4);
            au[3] = PICK(S1, p + 0);
            au[4] = PICK(S1, p + 2);
            au[5] = PICK(S1, p + 4);
            au[6] = PICK(S2, p + 0);
            au[7] = PICK(S2, p + 2);
            v16h a = __builtin_bit_cast(v16h, au);

            v8f acc = {};
            acc = __builtin_amdgcn_wmma_f32_16x16x32_f16(
                false, a, false, bfr[p], (short)0, acc, false, false);

            *(v8f*)&op[p * 8] = acc;                 // contiguous 32B: 2x ds_store_b128
        }
        __syncthreads();

        // ---- coalesced writeback: fixed (c, i) -> 16 consecutive output floats ----
        const int pj = lane & 15;
        const int pp = pj >> 1, jj = pj & 1;
        float* outp = out + ((long)(b * CB + c0) * (HB * UPF) + 2 * h) * (WB * UPF)
                    + (2 * wcol0 + pj);              // per-lane base; rest immediates
        #pragma unroll
        for (int it = 0; it < 16; ++it) {
            int i  = it & 1;
            int cl = (it >> 1) * 2 + khalf;
            int uu = i * UPF + jj;
            float v = u.ostage[wave][(uu * 2 + (cl >> 3)) * OGS + pp * 8 + (cl & 7)];
            outp[(long)cl * (HB * UPF) * (WB * UPF) + i * (WB * UPF)] = v;
        }
        __syncthreads();
    }
}

extern "C" void kernel_launch(void* const* d_in, const int* in_sizes, int n_in,
                              void* d_out, int out_size, void* d_ws, size_t ws_size,
                              hipStream_t stream) {
    const float* x   = (const float*)d_in[0];
    const float* km  = (const float*)d_in[1];
    float*       out = (float*)d_out;

    const int Bn = in_sizes[0] / (CB * HB * WB);     // 8
    dim3 grid(Bn * HB * (WB / PIX));                 // 4096 workgroups
    carafe_wmma_kernel<<<grid, THREADS, 0, stream>>>(x, km, out);
}
